// Quantiser_25709674234598
// MI455X (gfx1250) — matrix operand order repaired and checked by reading, not compiled, not run on hardware
//
#include <hip/hip_runtime.h>
#include <hip/hip_bf16.h>

#define VOCAB 8192
#define DIM   256
#define NTOK  32768
#define NTILE (VOCAB / 16)        // 512 code tiles of 16
#define MBLK  128                 // tokens per workgroup
#define TILE_BYTES (8 * 32 * 64)  // 8 kc-chunks * 32 lanes * 64B = 16 KB per code tile
#define NBUF  4                   // LDS pipeline depth

typedef __bf16 bfel;
typedef bfel  v16bf __attribute__((ext_vector_type(16)));
typedef float v8f   __attribute__((ext_vector_type(8)));
typedef unsigned int v8u __attribute__((ext_vector_type(8)));

static_assert(sizeof(v16bf) == 32, "v16bf must be 32 bytes");
static_assert(sizeof(v8u) == 32, "v8u must be 32 bytes");

// ---- bf16 helpers (bit-level, round-to-nearest for hi limb) -----------------
static __device__ __forceinline__ unsigned short bf_rn(float f) {
  unsigned u = __float_as_uint(f);
  u += 0x7fffu + ((u >> 16) & 1u);
  return (unsigned short)(u >> 16);
}
static __device__ __forceinline__ float bf_f(unsigned short h) {
  return __uint_as_float(((unsigned)h) << 16);
}

// ---- CDNA5 async global->LDS copy (ASYNCcnt path) ---------------------------
static __device__ __forceinline__ void async_ld16(void* ldsPtr, const void* gPtr) {
  // generic shared pointer: low 32 bits == LDS byte address (ISA aperture rule)
  unsigned lds_addr = (unsigned)(unsigned long long)ldsPtr;
  asm volatile("global_load_async_to_lds_b128 %0, %1, off"
               :: "v"(lds_addr), "v"(gPtr)
               : "memory");
}
static __device__ __forceinline__ void wait_async8() {
#if __has_builtin(__builtin_amdgcn_s_wait_asynccnt)
  __builtin_amdgcn_s_wait_asynccnt(8);
#else
  asm volatile("s_wait_asynccnt 8" ::: "memory");
#endif
}

// ============================================================================
// Kernel 1: split emb into bf16 hi/lo, pre-swizzled into per-lane B fragments;
//           also compute e2[v] = ||emb_v||^2.
// Fragment layout (per code tile vt, k-chunk kc, lane l):
//   code = 16*vt + (l&15);  K = 32*kc + 16*(l>>4) + j, j=0..15 (packed 2/dword)
//   storage: wsB[((vt*8+kc)*32+l)*32 + {0..15 hi, 16..31 lo}]  (u16 units)
// ============================================================================
__global__ void vq_prep(const float* __restrict__ emb,
                        unsigned short* __restrict__ wsB,
                        float* __restrict__ e2) {
  int tid = blockIdx.x * blockDim.x + threadIdx.x;  // 131072 threads
  int lane = tid & 31;
  int vt = tid >> 8;
  int kc = (tid >> 5) & 7;
  int code = vt * 16 + (lane & 15);
  int k0 = kc * 32 + ((lane >> 4) << 4);
  const float* row = emb + (size_t)code * DIM + k0;
  unsigned short* dst = wsB + (size_t)tid * 32;
#pragma unroll
  for (int j = 0; j < 16; ++j) {
    float f = row[j];
    unsigned short h = bf_rn(f);
    dst[j] = h;
    dst[16 + j] = bf_rn(f - bf_f(h));
  }
  if (tid < VOCAB) {
    const float4* r4 = (const float4*)(emb + (size_t)tid * DIM);
    float s = 0.f;
#pragma unroll 4
    for (int j = 0; j < DIM / 4; ++j) {
      float4 v = r4[j];
      s += v.x * v.x + v.y * v.y + v.z * v.z + v.w * v.w;
    }
    e2[tid] = s;
  }
}

// ============================================================================
// Kernel 2: fused GEMM (bf16x3 WMMA) + argmin + gather + partial MSE.
// 256 threads = 8 wave32s; wave w owns tokens [t0+16w, t0+16w+16).
// A fragments register-resident; B tiles flow through a 4-deep LDS pipeline
// fed by async global->LDS DMA; one barrier per tile.
// ============================================================================
__global__ __launch_bounds__(256)
void vq_main(const float* __restrict__ x,
             const float* __restrict__ emb,
             const unsigned short* __restrict__ wsB,
             const float* __restrict__ e2g,
             float* __restrict__ out_q,
             float* __restrict__ out_idx,
             float* __restrict__ partials) {
  __shared__ __align__(32) unsigned char ldsb[NBUF][TILE_BYTES];  // 64 KB
  __shared__ float lds_e2[VOCAB];                                 // 32 KB
  __shared__ float red[8];

  const int tid = threadIdx.x;
  const int lane = tid & 31;
  const int wave = tid >> 5;
  const int t0 = blockIdx.x * MBLK;
  const int mrow = lane & 15;
  const int khalf = (lane >> 4) << 4;  // 0 or 16: A-fragment K half
  const int token = t0 + wave * 16 + mrow;

  // ---- stage e2 into LDS (8 x b128 per thread) ----
#pragma unroll
  for (int i = 0; i < 8; ++i)
    ((float4*)lds_e2)[i * 256 + tid] = ((const float4*)e2g)[i * 256 + tid];

  // ---- prefetch first 3 code tiles into the LDS pipeline ----
#pragma unroll
  for (int p = 0; p < NBUF - 1; ++p) {
    const unsigned char* g = (const unsigned char*)wsB + (size_t)p * TILE_BYTES;
#pragma unroll
    for (int i = 0; i < 4; ++i) {
      int off = (i * 256 + tid) * 16;
      async_ld16(&ldsb[p][off], g + off);
    }
  }

  // ---- build register-resident A fragments (hi/lo bf16 split) ----
  // A 16x32 in contiguous K order (matches wsB packing): lanes 0-15 -> M=lane, K=0..15; lanes 16-31
  // -> M=lane-16, K=16..31; elements packed 2 per dword.
  v16bf a_hi[8], a_lo[8];
  const float* xrow = x + (size_t)token * DIM;
#pragma unroll
  for (int kc = 0; kc < 8; ++kc) {
    float f[16] __attribute__((aligned(16)));
    const float4* p0 = (const float4*)(xrow + kc * 32 + khalf);
    const float4* p1 = (const float4*)(xrow + kc * 32 + khalf + 8);
    ((float4*)f)[0] = p0[0];
    ((float4*)f)[1] = p0[1];
    ((float4*)f)[2] = p1[0];
    ((float4*)f)[3] = p1[1];
    v8u hi, lo;
#pragma unroll
    for (int w = 0; w < 8; ++w) {
      unsigned short h0 = bf_rn(f[2 * w]);
      unsigned short h1 = bf_rn(f[2 * w + 1]);
      unsigned short l0 = bf_rn(f[2 * w] - bf_f(h0));
      unsigned short l1 = bf_rn(f[2 * w + 1] - bf_f(h1));
      hi[w] = (unsigned)h0 | ((unsigned)h1 << 16);
      lo[w] = (unsigned)l0 | ((unsigned)l1 << 16);
    }
    a_hi[kc] = __builtin_bit_cast(v16bf, hi);
    a_lo[kc] = __builtin_bit_cast(v16bf, lo);
  }

  float bestv[8];
  int besti[8];
#pragma unroll
  for (int r = 0; r < 8; ++r) { bestv[r] = 3.402823466e38f; besti[r] = 0; }

  // ---- sweep all 512 code tiles; single barrier per tile ----
  for (int vt = 0; vt < NTILE; ++vt) {
    // 3 prefetches (4 async ops each) in flight; <=8 outstanding => oldest done
    wait_async8();
    __syncthreads();  // tile vt visible to all waves (incl. e2 stores on vt==0)

    {  // keep exactly 3 prefetches in flight (wrap-around keeps count uniform)
      const int nt = (vt + NBUF - 1) & (NTILE - 1);
      unsigned char* dst = ldsb[(vt + NBUF - 1) & (NBUF - 1)];
      const unsigned char* g =
          (const unsigned char*)wsB + (size_t)nt * TILE_BYTES;
#pragma unroll
      for (int i = 0; i < 4; ++i) {
        int off = (i * 256 + tid) * 16;
        async_ld16(dst + off, g + off);
      }
    }

    const unsigned char* bbase = ldsb[vt & (NBUF - 1)];
    v8f acc = {};
#pragma unroll
    for (int kc = 0; kc < 8; ++kc) {
      const v8u* bp = (const v8u*)&bbase[(kc * 32 + lane) * 64];
      v16bf bhi = __builtin_bit_cast(v16bf, bp[0]);
      v16bf blo = __builtin_bit_cast(v16bf, bp[1]);
      acc = __builtin_amdgcn_wmma_f32_16x16x32_bf16(false, a_hi[kc], false, bhi,
                                                    (short)0, acc, false, false);
      acc = __builtin_amdgcn_wmma_f32_16x16x32_bf16(false, a_hi[kc], false, blo,
                                                    (short)0, acc, false, false);
      acc = __builtin_amdgcn_wmma_f32_16x16x32_bf16(false, a_lo[kc], false, bhi,
                                                    (short)0, acc, false, false);
    }

    // score = ||e||^2 - 2 x.e  (||x||^2 dropped: constant per token)
    const int code = vt * 16 + mrow;
    const float ev = lds_e2[code];
#pragma unroll
    for (int r = 0; r < 8; ++r) {
      float s = fmaf(-2.0f, acc[r], ev);
      if (s < bestv[r]) { bestv[r] = s; besti[r] = code; }
    }
  }

  // ---- argmin across the 16 lanes of each C-row (first-index tie-break) ----
#pragma unroll
  for (int r = 0; r < 8; ++r) {
#pragma unroll
    for (int s = 1; s < 16; s <<= 1) {
      float ov = __shfl_xor(bestv[r], s, 32);
      int oi = __shfl_xor(besti[r], s, 32);
      if (ov < bestv[r] || (ov == bestv[r] && oi < besti[r])) {
        bestv[r] = ov;
        besti[r] = oi;
      }
    }
  }

  // C layout: VGPR r holds row M=r (lanes 0-15) / M=r+8 (lanes 16-31)
  int rowidx[16];
#pragma unroll
  for (int m = 0; m < 16; ++m)
    rowidx[m] = __shfl(besti[m & 7], (m >> 3) * 16, 32);

  if (lane == 0) {
#pragma unroll
    for (int m = 0; m < 16; ++m)
      out_idx[t0 + wave * 16 + m] = (float)rowidx[m];
  }

  // ---- gather quantised rows, straight-through output, local MSE ----
  float sq = 0.f;
#pragma unroll
  for (int m = 0; m < 16; ++m) {
    int tok = t0 + wave * 16 + m;
    const float4* er = (const float4*)(emb + (size_t)rowidx[m] * DIM) + lane * 2;
    const float4* xr = (const float4*)(x + (size_t)tok * DIM) + lane * 2;
    float4* qr = (float4*)(out_q + (size_t)tok * DIM) + lane * 2;
#pragma unroll
    for (int j = 0; j < 2; ++j) {
      float4 e4 = er[j];
      float4 x4 = xr[j];
      float dx = e4.x - x4.x, dy = e4.y - x4.y;
      float dz = e4.z - x4.z, dw = e4.w - x4.w;
      sq += dx * dx + dy * dy + dz * dz + dw * dw;
      float4 o;
      o.x = x4.x + dx; o.y = x4.y + dy; o.z = x4.z + dz; o.w = x4.w + dw;
      qr[j] = o;
    }
  }
#pragma unroll
  for (int s = 1; s < 32; s <<= 1) sq += __shfl_xor(sq, s, 32);
  if (lane == 0) red[wave] = sq;
  __syncthreads();
  if (tid == 0) {
    float t = 0.f;
#pragma unroll
    for (int w = 0; w < 8; ++w) t += red[w];
    partials[blockIdx.x] = t;  // plain store: deterministic, no atomics
  }
}

// ============================================================================
// Kernel 3: deterministic final reduction -> loss scalars.
// codebook_loss == e_latent_loss numerically -> loss = 1.25 * codebook_loss.
// ============================================================================
__global__ void vq_finish(const float* __restrict__ partials,
                          float* __restrict__ out_scalars) {
  if (blockIdx.x == 0 && threadIdx.x == 0) {
    float s = 0.f;
    for (int i = 0; i < NTOK / MBLK; ++i) s += partials[i];
    float cb = s * (1.0f / (float)(NTOK * DIM));
    out_scalars[0] = cb + 0.25f * cb;  // loss
    out_scalars[1] = cb;               // codebook_loss
  }
}

// ============================================================================
extern "C" void kernel_launch(void* const* d_in, const int* in_sizes, int n_in,
                              void* d_out, int out_size, void* d_ws,
                              size_t ws_size, hipStream_t stream) {
  const float* x = (const float*)d_in[0];     // [8,4096,256] f32
  const float* emb = (const float*)d_in[1];   // [8192,256]  f32

  float* out = (float*)d_out;
  float* out_q = out;                              // 8388608 floats
  float* out_scalars = out + (size_t)NTOK * DIM;   // loss, codebook_loss
  float* out_idx = out + (size_t)NTOK * DIM + 2;   // 32768 indices (as float)

  // workspace: 8MB swizzled bf16 hi/lo codebook, 32KB e2, 1KB partials
  unsigned short* wsB = (unsigned short*)d_ws;
  float* e2 = (float*)((char*)d_ws + (size_t)VOCAB * DIM * 2 * sizeof(unsigned short));
  float* partials = e2 + VOCAB;

  vq_prep<<<(VOCAB / 16) * 8 * 32 / 256, 256, 0, stream>>>(emb, wsB, e2);
  vq_main<<<NTOK / MBLK, 256, 0, stream>>>(x, emb, wsB, e2, out_q, out_idx,
                                           partials);
  vq_finish<<<1, 64, 0, stream>>>(partials, out_scalars);
}
